// GMHCA_85212151152821
// MI455X (gfx1250) — compile-verified, matrix operand order checked by reading, not run on hardware
//
#include <hip/hip_runtime.h>
#include <hip/hip_bf16.h>

// ---------------------------------------------------------------------------
// Multi-head cross-attention for MI455X (gfx1250, wave32, WMMA).
// All GEMMs use v_wmma_f32_16x16x32_bf16 (bf16 operands, f32 accumulate).
// SCALE (=1/8, exact) is folded into WqT so the scores kernel needs no
// per-element multiply. Workspace layout (requires ws_size >= 80 MB):
//   [ 0MB)  WqT  bf16 [1024][1024]   (pre-scaled by 0.125)
//   [ 2MB)  WkvT bf16 [2048][1024]
//   [ 6MB)  WoT  bf16 [1024][1024]
//   [ 8MB)  q    bf16 [4][1024][1024]
//   [16MB)  kv   bf16 [4][2048][2048]   (k = cols 0..1023, v = cols 1024..2047)
//   [48MB)  vT   bf16 [4][16][64][2048]
//   [64MB)  ctx  f32  [4][1024][1024]
// d_out: out f32 [4][1024][1024] (4,194,304) then attn f32 [64][1024][2048].
// ---------------------------------------------------------------------------

typedef __attribute__((ext_vector_type(16))) __bf16 v16bf;
typedef __attribute__((ext_vector_type(2)))  __bf16 v2bf;
typedef __attribute__((ext_vector_type(16))) float  v16f;
typedef __attribute__((ext_vector_type(8)))  float  v8f;

#define WMMA_BF16(a, b, c) \
    __builtin_amdgcn_wmma_f32_16x16x32_bf16(false, (a), false, (b), (short)0, (c), false, false)

static __device__ __forceinline__ v16bf load_cvt16(const float* __restrict__ p) {
    v16f f = *(const v16f*)p;
    v16bf r;
#if __has_builtin(__builtin_amdgcn_cvt_pk_bf16_f32)
#pragma unroll
    for (int i = 0; i < 8; ++i) {
        v2bf pk = __builtin_amdgcn_cvt_pk_bf16_f32(f[2 * i], f[2 * i + 1]);
        r[2 * i]     = pk[0];
        r[2 * i + 1] = pk[1];
    }
#else
#pragma unroll
    for (int i = 0; i < 16; ++i) r[i] = (__bf16)f[i];
#endif
    return r;
}

// ---------------------------------------------------------------------------
// WT[n][k] = (bf16)(W[k][n] * scale)   (W is [K][N] row-major)
// ---------------------------------------------------------------------------
__global__ void transpose_cvt_kernel(const float* __restrict__ W, __bf16* __restrict__ WT,
                                     int K, int N, float scale) {
    int idx = blockIdx.x * blockDim.x + threadIdx.x;
    if (idx >= K * N) return;
    int n = idx / K;
    int k = idx % K;
    WT[idx] = (__bf16)(W[(size_t)k * N + n] * scale);
}

// ---------------------------------------------------------------------------
// vT[b][h][d][j] = kv[b][j][1024 + h*64 + d]
// ---------------------------------------------------------------------------
__global__ void vtrans_kernel(const __bf16* __restrict__ kv, __bf16* __restrict__ vT) {
    int idx = blockIdx.x * blockDim.x + threadIdx.x;
    if (idx >= 4 * 16 * 64 * 2048) return;
    int j = idx & 2047;
    int t = idx >> 11;
    int d = t & 63;  t >>= 6;
    int h = t & 15;
    int b = t >> 4;
    vT[idx] = kv[((size_t)b * 2048 + j) * 2048 + 1024 + h * 64 + d];
}

// ---------------------------------------------------------------------------
// Generic GEMM: C[M][N] = A[M][K] (f32, cvt->bf16) * BT[N][K] (bf16)
// Block = 128 threads (4 waves). Each wave covers 32 rows x 64 cols
// (2x4 WMMA tiles, 8 accumulators; B fragments reused across 2 M tiles).
// grid = (N/256, M/32).
// ---------------------------------------------------------------------------
template <bool F32OUT>
__global__ void gemm_wmma_kernel(const float* __restrict__ A, int lda,
                                 const __bf16* __restrict__ BT,
                                 void* __restrict__ Cp, int ldc,
                                 const float* __restrict__ bias, int K) {
    const int lane = threadIdx.x & 31;
    const int wid  = threadIdx.x >> 5;
    const int m0   = blockIdx.y * 32;
    const int n0   = (blockIdx.x * 4 + wid) * 64;
    const int col  = lane & 15;
    const int half = lane >> 4;

    v8f acc[2][4] = {};
    const float* arow0 = A + (size_t)(m0 + col) * lda + half * 16;
    const float* arow1 = arow0 + (size_t)16 * lda;
    for (int kt = 0; kt < K; kt += 32) {
        v16bf a0 = load_cvt16(arow0 + kt);
        v16bf a1 = load_cvt16(arow1 + kt);
#pragma unroll
        for (int nb = 0; nb < 4; ++nb) {
            const __bf16* bp = BT + (size_t)(n0 + nb * 16 + col) * K + kt + half * 16;
            v16bf b = *(const v16bf*)bp;
            acc[0][nb] = WMMA_BF16(a0, b, acc[0][nb]);
            acc[1][nb] = WMMA_BF16(a1, b, acc[1][nb]);
        }
    }
#pragma unroll
    for (int mt = 0; mt < 2; ++mt) {
#pragma unroll
        for (int nb = 0; nb < 4; ++nb) {
            const int c = n0 + nb * 16 + col;
            if (F32OUT) {
                const float bv = bias ? bias[c] : 0.0f;
                float* C = (float*)Cp;
#pragma unroll
                for (int r = 0; r < 8; ++r)
                    C[(size_t)(m0 + mt * 16 + r + half * 8) * ldc + c] = acc[mt][nb][r] + bv;
            } else {
                __bf16* C = (__bf16*)Cp;
#pragma unroll
                for (int r = 0; r < 8; ++r)
                    C[(size_t)(m0 + mt * 16 + r + half * 8) * ldc + c] = (__bf16)acc[mt][nb][r];
            }
        }
    }
}

// ---------------------------------------------------------------------------
// Scores + softmax: attn[b*16+h][i][j] = softmax_j( q_h . k_h )
// (q already carries the 1/sqrt(d) scale.)
// One wave per (b, h, 16-row q tile). 64 kv columns per iteration
// (4 accumulators, 8 WMMAs); row max/sum combined across the 4 tiles with
// VALU before a single 16-lane butterfly. Raw scores streamed to d_out,
// then in-place normalize (L2-hot).
// grid = (Lq/64, 16, 4), block = 128.
// ---------------------------------------------------------------------------
__global__ void attn_scores_kernel(const __bf16* __restrict__ q,   // [4][1024][1024]
                                   const __bf16* __restrict__ kv,  // [4][2048][2048]
                                   float* __restrict__ attn) {     // [64][1024][2048]
    const int Lq = 1024, Lkv = 2048;
    const int lane = threadIdx.x & 31;
    const int wid  = threadIdx.x >> 5;
    const int qt   = blockIdx.x * 4 + wid;
    const int h    = blockIdx.y;
    const int b    = blockIdx.z;
    const int i0   = qt * 16;
    const int col  = lane & 15;
    const int half = lane >> 4;

    const __bf16* qrow = q + ((size_t)b * Lq + i0 + col) * 1024 + h * 64 + half * 16;
    const v16bf a0 = *(const v16bf*)qrow;
    const v16bf a1 = *(const v16bf*)(qrow + 32);

    float* arow = attn + ((size_t)(b * 16 + h) * Lq + i0) * Lkv;

    float mrun[8], lrun[8];
#pragma unroll
    for (int r = 0; r < 8; ++r) { mrun[r] = -1e30f; lrun[r] = 0.0f; }

    for (int j0 = 0; j0 < Lkv; j0 += 64) {
        v8f acc[4];
#pragma unroll
        for (int t = 0; t < 4; ++t) {
            const __bf16* krow =
                kv + ((size_t)b * Lkv + j0 + t * 16 + col) * 2048 + h * 64 + half * 16;
            v16bf b0 = *(const v16bf*)krow;
            v16bf b1 = *(const v16bf*)(krow + 32);
            v8f z = {};
            z = WMMA_BF16(a0, b0, z);
            acc[t] = WMMA_BF16(a1, b1, z);
        }

        // per-row max over the 4 tiles (VALU), then one 16-lane butterfly
        float red[8];
#pragma unroll
        for (int r = 0; r < 8; ++r)
            red[r] = fmaxf(fmaxf(acc[0][r], acc[1][r]), fmaxf(acc[2][r], acc[3][r]));
#pragma unroll
        for (int msk = 8; msk >= 1; msk >>= 1)
#pragma unroll
            for (int r = 0; r < 8; ++r)
                red[r] = fmaxf(red[r], __shfl_xor(red[r], msk, 16));

        // online running-max update (once per 64 columns)
#pragma unroll
        for (int r = 0; r < 8; ++r) {
            const float mnew = fmaxf(mrun[r], red[r]);
            lrun[r] *= __expf(mrun[r] - mnew);
            mrun[r] = mnew;
        }
        // exp + partial sums (VALU across tiles), one butterfly sum
        float psum[8];
#pragma unroll
        for (int r = 0; r < 8; ++r) psum[r] = 0.0f;
#pragma unroll
        for (int t = 0; t < 4; ++t)
#pragma unroll
            for (int r = 0; r < 8; ++r) psum[r] += __expf(acc[t][r] - mrun[r]);
#pragma unroll
        for (int msk = 8; msk >= 1; msk >>= 1)
#pragma unroll
            for (int r = 0; r < 8; ++r) psum[r] += __shfl_xor(psum[r], msk, 16);
#pragma unroll
        for (int r = 0; r < 8; ++r) lrun[r] += psum[r];

        // stream raw scores
#pragma unroll
        for (int t = 0; t < 4; ++t)
#pragma unroll
            for (int r = 0; r < 8; ++r)
                arow[(size_t)(r + half * 8) * Lkv + j0 + t * 16 + col] = acc[t][r];
    }

    float inv[8];
#pragma unroll
    for (int r = 0; r < 8; ++r) inv[r] = 1.0f / lrun[r];
    // normalize in place (same-wave store->load ordering is guaranteed)
    for (int j0 = 0; j0 < Lkv; j0 += 16) {
#pragma unroll
        for (int r = 0; r < 8; ++r) {
            const size_t off = (size_t)(r + half * 8) * Lkv + j0 + col;
            arow[off] = __expf(arow[off] - mrun[r]) * inv[r];
        }
    }
}

// ---------------------------------------------------------------------------
// ctx[b][i][h*64+d] = sum_j attn[b*16+h][i][j] * v[b][j][h*64+d]
// One wave per (b, h, 32-row q tile) computes 32 x 64 (8 accumulators;
// vT fragments reused across both M tiles).
// grid = (Lq/128, 16, 4), block = 128.
// ---------------------------------------------------------------------------
__global__ void attn_v_kernel(const float* __restrict__ attn,  // [64][1024][2048]
                              const __bf16* __restrict__ vT,   // [4][16][64][2048]
                              float* __restrict__ ctx) {       // [4][1024][1024]
    const int Lq = 1024, Lkv = 2048;
    const int lane = threadIdx.x & 31;
    const int wid  = threadIdx.x >> 5;
    const int qt   = blockIdx.x * 4 + wid;
    const int h    = blockIdx.y;
    const int b    = blockIdx.z;
    const int i0   = qt * 32;
    const int col  = lane & 15;
    const int half = lane >> 4;

    const float* prow0  = attn + ((size_t)(b * 16 + h) * Lq + i0 + col) * Lkv + half * 16;
    const float* prow1  = prow0 + (size_t)16 * Lkv;
    const __bf16* vbase = vT + ((size_t)b * 16 + h) * 64 * 2048;

    v8f acc[2][4] = {};
    for (int j0 = 0; j0 < Lkv; j0 += 32) {
        v16bf a0 = load_cvt16(prow0 + j0);
        v16bf a1 = load_cvt16(prow1 + j0);
#pragma unroll
        for (int db = 0; db < 4; ++db) {
            const __bf16* vp = vbase + (size_t)(db * 16 + col) * 2048 + j0 + half * 16;
            v16bf bf = *(const v16bf*)vp;
            acc[0][db] = WMMA_BF16(a0, bf, acc[0][db]);
            acc[1][db] = WMMA_BF16(a1, bf, acc[1][db]);
        }
    }
    float* crow = ctx + ((size_t)b * Lq + i0) * 1024 + h * 64;
#pragma unroll
    for (int mt = 0; mt < 2; ++mt)
#pragma unroll
        for (int db = 0; db < 4; ++db)
#pragma unroll
            for (int r = 0; r < 8; ++r)
                crow[(size_t)(mt * 16 + r + half * 8) * 1024 + db * 16 + col] = acc[mt][db][r];
}

// ---------------------------------------------------------------------------
extern "C" void kernel_launch(void* const* d_in, const int* in_sizes, int n_in,
                              void* d_out, int out_size, void* d_ws, size_t ws_size,
                              hipStream_t stream) {
    const float* query   = (const float*)d_in[0];  // [4][1024][1024]
    const float* context = (const float*)d_in[1];  // [4][2048][1024]
    const float* Wq      = (const float*)d_in[2];  // [1024][1024]
    const float* Wkv     = (const float*)d_in[3];  // [1024][2048]
    const float* Wo      = (const float*)d_in[4];  // [1024][1024]
    const float* bo      = (const float*)d_in[5];  // [1024]

    char* ws = (char*)d_ws;
    const size_t MB = 1024 * 1024;
    __bf16* WqT  = (__bf16*)(ws + 0 * MB);
    __bf16* WkvT = (__bf16*)(ws + 2 * MB);
    __bf16* WoT  = (__bf16*)(ws + 6 * MB);
    __bf16* qb   = (__bf16*)(ws + 8 * MB);
    __bf16* kvb  = (__bf16*)(ws + 16 * MB);
    __bf16* vT   = (__bf16*)(ws + 48 * MB);
    float*  ctx  = (float*)(ws + 64 * MB);

    float* out_p  = (float*)d_out;                            // [4][1024][1024]
    float* attn_p = (float*)d_out + (size_t)4 * 1024 * 1024;  // [64][1024][2048]

    // 1) Weight transposes (f32 -> bf16, [K][N] -> [N][K]).
    //    Wq carries the 1/sqrt(64) attention scale (exact power of two).
    transpose_cvt_kernel<<<(1024 * 1024 + 255) / 256, 256, 0, stream>>>(Wq, WqT, 1024, 1024, 0.125f);
    transpose_cvt_kernel<<<(1024 * 2048 + 255) / 256, 256, 0, stream>>>(Wkv, WkvT, 1024, 2048, 1.0f);
    transpose_cvt_kernel<<<(1024 * 1024 + 255) / 256, 256, 0, stream>>>(Wo, WoT, 1024, 1024, 1.0f);

    // 2) q = query @ (Wq*scale)   (M=4096, N=1024, K=1024)
    gemm_wmma_kernel<false><<<dim3(1024 / 256, 4096 / 32), 128, 0, stream>>>(
        query, 1024, WqT, qb, 1024, nullptr, 1024);
    //    kv = context @ Wkv  (M=8192, N=2048, K=1024)
    gemm_wmma_kernel<false><<<dim3(2048 / 256, 8192 / 32), 128, 0, stream>>>(
        context, 1024, WkvT, kvb, 2048, nullptr, 1024);

    // 3) v -> vT
    vtrans_kernel<<<(4 * 16 * 64 * 2048 + 255) / 256, 256, 0, stream>>>(kvb, vT);

    // 4) attention scores + softmax (writes attn probs to d_out)
    attn_scores_kernel<<<dim3(1024 / 64, 16, 4), 128, 0, stream>>>(qb, kvb, attn_p);

    // 5) ctx = attn @ v
    attn_v_kernel<<<dim3(1024 / 128, 16, 4), 128, 0, stream>>>(attn_p, vT, ctx);

    // 6) out = ctx @ Wo + bo  (M=4096, N=1024, K=1024, f32 out)
    gemm_wmma_kernel<true><<<dim3(1024 / 256, 4096 / 32), 128, 0, stream>>>(
        ctx, 1024, WoT, out_p, 1024, bo, 1024);
}
